// Conv1DRnn_49469433316058
// MI455X (gfx1250) — matrix operand order repaired and checked by reading, not compiled, not run on hardware
//
#include <hip/hip_runtime.h>

// Problem constants (from reference): B=16, T=64, C=128, H=256, L=128, K=3
#define B_ 16
#define T_ 64
#define C_ 128
#define H_ 256
#define L_ 128

typedef __attribute__((ext_vector_type(16))) __bf16 v16bf;
typedef __attribute__((ext_vector_type(8)))  float  v8f;
typedef __attribute__((ext_vector_type(4)))  unsigned int u32x4;

// Pointer-element type expected by the async-to-LDS builtins (from hipcc
// diagnostic: "int __attribute__((vector_size(16))) <AS> *").
typedef __attribute__((__vector_size__(16))) int vs4i;
typedef __attribute__((address_space(1))) vs4i* gptr_v4i;
typedef __attribute__((address_space(3))) vs4i* lptr_v4i;

union Frag16 {          // 16 bf16 halves == 8 VGPRs; filled via two 16B loads
    v16bf v;
    u32x4 q[2];
};

// gfx1250 async global->LDS path (ASYNCcnt-tracked), with safe fallback.
#if defined(__has_builtin)
#if __has_builtin(__builtin_amdgcn_global_load_async_to_lds_b128) && \
    __has_builtin(__builtin_amdgcn_s_wait_asynccnt)
#define USE_ASYNC_LDS 1
#endif
#endif

__device__ __forceinline__ unsigned short f2bf(float f) {
    unsigned int u = __float_as_uint(f);
    u += 0x7FFFu + ((u >> 16) & 1u);    // round-to-nearest-even
    return (unsigned short)(u >> 16);
}

// ---------------------------------------------------------------------------
// Weight repack: W[H,Cin,K] f32  ->  Wb[K][H][Cin] bf16 (A rows contiguous)
// ---------------------------------------------------------------------------
__global__ void conv1drnn_convert_weights(const float* __restrict__ Wi,
                                          const float* __restrict__ Wh,
                                          unsigned short* __restrict__ WiB,
                                          unsigned short* __restrict__ WhB) {
    int idx = blockIdx.x * blockDim.x + threadIdx.x;
    if (idx < 3 * H_ * C_) {
        int k = idx / (H_ * C_);
        int r = idx - k * (H_ * C_);
        int h = r / C_, c = r - h * C_;
        WiB[idx] = f2bf(Wi[(h * C_ + c) * 3 + k]);
    }
    if (idx < 3 * H_ * H_) {
        int k = idx / (H_ * H_);
        int r = idx - k * (H_ * H_);
        int h = r / H_, c = r - h * H_;
        WhB[idx] = f2bf(Wh[(h * H_ + c) * 3 + k]);
    }
}

__global__ void conv1drnn_zero_u16(unsigned short* __restrict__ p, int n) {
    int i = blockIdx.x * blockDim.x + threadIdx.x;
    if (i < n) p[i] = 0;
}

// ---------------------------------------------------------------------------
// Phase 1 (fully parallel): U = conv_i(x_t) for ALL (b,t) at once, written
// straight into d_out[b,t,:,:].
// grid = (L/32, H/64, B*T), block = 256 (8 waves, one 16x16 tile each)
// ---------------------------------------------------------------------------
__global__ __launch_bounds__(256) void conv1drnn_input_proj(
    const float*          __restrict__ x,     // [B*T, C, L] f32
    const unsigned short* __restrict__ WiB,   // [3,H,C] bf16
    float*                __restrict__ out)   // [B*T, H, L] f32
{
    __shared__ alignas(16) unsigned short ldsX[34 * 136];

    const int tid = threadIdx.x;
    const int l0  = blockIdx.x * 32;
    const int h0  = blockIdx.y * 64;
    const int bt  = blockIdx.z;

    const float* xt = x + (size_t)bt * C_ * L_;
    for (int i = tid; i < 34 * C_; i += 256) {
        int c = i / 34, col = i - c * 34;
        int l = l0 - 1 + col;
        float v = (l >= 0 && l < L_) ? xt[c * L_ + l] : 0.0f;
        ldsX[col * 136 + c] = f2bf(v);
    }
    __syncthreads();

    const int wave = tid >> 5, lane = tid & 31;
    const int wh   = wave >> 1, wl = wave & 1;
    const int m    = lane & 15, sel = lane >> 4;
    const int hrow = h0 + wh * 16 + m;

    v8f acc = {};
    #pragma unroll
    for (int k = 0; k < 3; ++k) {
        const unsigned short* wrow = WiB + ((size_t)k * H_ + hrow) * C_;
        const unsigned short* brow = &ldsX[(wl * 16 + m + k) * 136];
        #pragma unroll
        for (int kc = 0; kc < C_; kc += 32) {
            Frag16 a, bm;
            a.q[0] = *(const u32x4*)(wrow + kc + sel * 8);
            a.q[1] = *(const u32x4*)(wrow + kc + 16 + sel * 8);
            bm.q[0] = *(const u32x4*)(brow + kc + sel * 8);
            bm.q[1] = *(const u32x4*)(brow + kc + 16 + sel * 8);
            acc = __builtin_amdgcn_wmma_f32_16x16x32_bf16(
                false, a.v, false, bm.v, (short)0, acc, false, false);
        }
    }

    const int ncol = l0 + wl * 16 + m;
    float* op = out + (size_t)bt * H_ * L_;
    #pragma unroll
    for (int j = 0; j < 8; ++j)
        op[(size_t)(h0 + wh * 16 + sel * 8 + j) * L_ + ncol] = acc[j];
}

// ---------------------------------------------------------------------------
// Phase 2 (serial over t): h_t = tanh(U_t + conv_h(h_{t-1}))
// Recurrent state kept TRANSPOSED in global, [B][L][H] bf16, so LDS staging is
// contiguous 16B chunks -> global_load_async_to_lds_b128, and the state write
// is one packed b128 store per lane.
// grid = (L/32, H/64, B), block = 256
// ---------------------------------------------------------------------------
__global__ __launch_bounds__(256) void conv1drnn_step(
    const unsigned short* __restrict__ WhB,   // [3,H,H] bf16
    const unsigned short* __restrict__ hprev, // [B,L,H] bf16 (transposed)
    unsigned short*       __restrict__ hcur,  // [B,L,H] bf16 (transposed)
    float*                __restrict__ out,   // [B,T,H,L] f32 (holds U, overwritten)
    int t)
{
    __shared__ alignas(16) unsigned short ldsH[34 * 264];   // [col][h'], padded pitch

    const int tid = threadIdx.x;
    const int l0  = blockIdx.x * 32;
    const int h0  = blockIdx.y * 64;
    const int b   = blockIdx.z;

    const int wave = tid >> 5, lane = tid & 31;
    const int wh   = wave >> 1, wl = wave & 1;
    const int m    = lane & 15, sel = lane >> 4;
    const int hrow = h0 + wh * 16 + m;
    const int ncol = l0 + wl * 16 + m;
    const int mrow = h0 + wh * 16 + sel * 8;

    // Kick off the U reads (independent of LDS staging).
    float* op = out + (((size_t)b * T_ + t) * H_) * L_;
    float init[8];
    #pragma unroll
    for (int j = 0; j < 8; ++j)
        init[j] = op[(size_t)(mrow + j) * L_ + ncol];

    // ---- stage h_{t-1}: 34 halo columns x 32 chunks of 16B each ----
    const unsigned short* hp = hprev + (size_t)b * L_ * H_;
    for (int i = tid; i < 34 * 32; i += 256) {
        int col = i >> 5, seg = i & 31;
        int l = l0 - 1 + col;
        unsigned short* ld = &ldsH[col * 264 + seg * 8];
        if (l >= 0 && l < L_) {
            const unsigned short* gp = hp + (size_t)l * H_ + seg * 8;
#ifdef USE_ASYNC_LDS
            __builtin_amdgcn_global_load_async_to_lds_b128(
                (gptr_v4i)(void*)gp, (lptr_v4i)(void*)ld, 0, 0);
#else
            *(u32x4*)ld = *(const u32x4*)gp;
#endif
        } else {
            u32x4 z = {0u, 0u, 0u, 0u};
            *(u32x4*)ld = z;
        }
    }
#ifdef USE_ASYNC_LDS
    __builtin_amdgcn_s_wait_asynccnt(0);
#endif
    __syncthreads();

    // Two accumulators -> two independent WMMA dependency chains.
    v8f acc0 = {}, acc1 = {};
    #pragma unroll
    for (int j = 0; j < 8; ++j) acc0[j] = init[j];

    #pragma unroll
    for (int k = 0; k < 3; ++k) {
        const unsigned short* wrow = WhB + ((size_t)k * H_ + hrow) * H_;
        const unsigned short* brow = &ldsH[(wl * 16 + m + k) * 264];
        #pragma unroll
        for (int kc = 0; kc < H_; kc += 64) {
            Frag16 a0, b0, a1, b1;
            a0.q[0] = *(const u32x4*)(wrow + kc + sel * 8);
            a0.q[1] = *(const u32x4*)(wrow + kc + 16 + sel * 8);
            b0.q[0] = *(const u32x4*)(brow + kc + sel * 8);
            b0.q[1] = *(const u32x4*)(brow + kc + 16 + sel * 8);
            acc0 = __builtin_amdgcn_wmma_f32_16x16x32_bf16(
                false, a0.v, false, b0.v, (short)0, acc0, false, false);

            a1.q[0] = *(const u32x4*)(wrow + kc + 32 + sel * 8);
            a1.q[1] = *(const u32x4*)(wrow + kc + 48 + sel * 8);
            b1.q[0] = *(const u32x4*)(brow + kc + 32 + sel * 8);
            b1.q[1] = *(const u32x4*)(brow + kc + 48 + sel * 8);
            acc1 = __builtin_amdgcn_wmma_f32_16x16x32_bf16(
                false, a1.v, false, b1.v, (short)0, acc1, false, false);
        }
    }

    // ---- epilogue: tanh, f32 output in-place over U, packed bf16 state ----
    alignas(16) unsigned short hv[8];
    #pragma unroll
    for (int j = 0; j < 8; ++j) {
        float v = tanhf(acc0[j] + acc1[j]);
        op[(size_t)(mrow + j) * L_ + ncol] = v;
        hv[j] = f2bf(v);
    }
    unsigned short* hc = hcur + ((size_t)b * L_ + ncol) * H_ + mrow;
    *(u32x4*)hc = *(const u32x4*)hv;
}

// ---------------------------------------------------------------------------
extern "C" void kernel_launch(void* const* d_in, const int* in_sizes, int n_in,
                              void* d_out, int out_size, void* d_ws, size_t ws_size,
                              hipStream_t stream) {
    const float* x  = (const float*)d_in[0];   // [B,T,C,L]
    const float* Wi = (const float*)d_in[1];   // [H,C,3]
    const float* Wh = (const float*)d_in[2];   // [H,H,3]
    float* out = (float*)d_out;                // [B,T,H,L]

    // Workspace (~2.6 MB, re-initialized every call):
    unsigned short* WiB = (unsigned short*)d_ws;        // 3*H*C bf16
    unsigned short* WhB = WiB + 3 * H_ * C_;            // 3*H*H bf16
    unsigned short* hA  = WhB + 3 * H_ * H_;            // B*L*H bf16 (state ping)
    unsigned short* hB  = hA + (size_t)B_ * L_ * H_;    // B*L*H bf16 (state pong)

    conv1drnn_convert_weights<<<(3 * H_ * H_ + 255) / 256, 256, 0, stream>>>(
        Wi, Wh, WiB, WhB);
    conv1drnn_zero_u16<<<(B_ * L_ * H_ + 255) / 256, 256, 0, stream>>>(
        hA, B_ * L_ * H_);

    // Phase 1: all input projections at once (full-machine parallel GEMM).
    conv1drnn_input_proj<<<dim3(L_ / 32, H_ / 64, B_ * T_), 256, 0, stream>>>(
        x, WiB, out);

    // Phase 2: serial recurrence, one small kernel per timestep.
    unsigned short* hprev = hA;
    unsigned short* hcur  = hB;
    dim3 grid(L_ / 32, H_ / 64, B_);
    for (int t = 0; t < T_; ++t) {
        conv1drnn_step<<<grid, 256, 0, stream>>>(WhB, hprev, hcur, out, t);
        unsigned short* tmp = hprev; hprev = hcur; hcur = tmp;
    }
}
